// Attention_33775622816043
// MI455X (gfx1250) — matrix operand order repaired and checked by reading, not compiled, not run on hardware
//
#include <hip/hip_runtime.h>
#include <hip/hip_bf16.h>

// ---------------------------------------------------------------------------
// Causal attention for MI455X (gfx1250, wave32, WMMA).
//   B=4, S=2048, d_model=1024, d_head=64.
// Strategy: bf16 WMMA (v_wmma_f32_16x16x32_bf16) with fp32 accumulation,
// flash-attention online softmax, no LDS (layout tricks keep everything in
// registers + shfl_xor(16)), all global traffic as 16B vectors, plus
// global_prefetch_b8 for the next kv-step to cover the load->WMMA chain.
// ---------------------------------------------------------------------------

typedef __attribute__((ext_vector_type(8)))  float  v8f;
typedef __attribute__((ext_vector_type(16))) __bf16 v16bf;

union ABits {
    v16bf          v;
    uint4          q[2];
    unsigned       u[8];
    unsigned short s[16];
};

static constexpr int kB  = 4;
static constexpr int kS  = 2048;
static constexpr int kDM = 1024;
static constexpr int kDH = 64;
static constexpr int kM  = kB * kS;          // 8192 tokens
static constexpr int kMTiles = kM / 16;      // 512

__device__ __forceinline__ unsigned short f2bf(float f) {
    __hip_bfloat16 h = __float2bfloat16(f);
    return *reinterpret_cast<unsigned short*>(&h);
}
__device__ __forceinline__ unsigned pack2(float a, float b) {
    return (unsigned)f2bf(a) | ((unsigned)f2bf(b) << 16);
}

// ---------------------------------------------------------------------------
// Kernel 0: Wt[m][n][k] = W_m[k][n] as bf16 (contiguous along K for WMMA B).
// ---------------------------------------------------------------------------
__global__ __launch_bounds__(256) void wt_kernel(const float* __restrict__ Wq,
                                                 const float* __restrict__ Wk,
                                                 const float* __restrict__ Wv,
                                                 unsigned short* __restrict__ wt) {
    int idx = blockIdx.x * 256 + threadIdx.x;      // 3*64*1024 = 196608 exact
    int m = idx >> 16;
    int r = idx & 65535;
    int n = r >> 10;
    int k = r & 1023;
    const float* W = (m == 0) ? Wq : ((m == 1) ? Wk : Wv);
    wt[idx] = f2bf(W[k * kDH + n]);
}

// ---------------------------------------------------------------------------
// Kernel 1: QKV projection. One wave per 16-token tile, all 3*64 outputs.
//   q,k row-major bf16 [B*S][64];  v stored transposed vT[B][64][S] bf16.
// ---------------------------------------------------------------------------
__global__ __launch_bounds__(256) void qkv_kernel(const float* __restrict__ x,
                                                  const unsigned short* __restrict__ wt,
                                                  const float* __restrict__ bq,
                                                  const float* __restrict__ bk,
                                                  const float* __restrict__ bv,
                                                  unsigned short* __restrict__ qb,
                                                  unsigned short* __restrict__ kb,
                                                  unsigned short* __restrict__ vtb) {
    const int lane = threadIdx.x & 31;
    const int lq   = lane & 15;          // column / row-in-group index
    const int h    = lane >> 4;          // half-wave group
    const int tile = blockIdx.x * 8 + (threadIdx.x >> 5);
    const int Mbase = tile * 16;

    const float* xrow = x + (size_t)(Mbase + lq) * kDM;

    v8f acc[3][4];
#pragma unroll
    for (int m = 0; m < 3; ++m)
#pragma unroll
        for (int nt = 0; nt < 4; ++nt)
            acc[m][nt] = (v8f){0.f, 0.f, 0.f, 0.f, 0.f, 0.f, 0.f, 0.f};

    for (int kk = 0; kk < kDM / 32; ++kk) {
        // Prefetch next A chunk (this wave streams xrow sequentially).
        if (kk + 1 < kDM / 32)
            __builtin_prefetch(xrow + (kk + 1) * 32 + 8 * h, 0, 3);

        // A operand: x chunk 16x32 f32 -> bf16, per ISA 16-bit A layout:
        //   elem i<8  : K = 8h + i        (contiguous 8 floats)
        //   elem i>=8 : K = 16 + 8h + i-8 (contiguous 8 floats)
        const float4* xc = (const float4*)(xrow + kk * 32);
        float4 a0 = xc[2 * h + 0];
        float4 a1 = xc[2 * h + 1];
        float4 a2 = xc[2 * h + 4];
        float4 a3 = xc[2 * h + 5];
        ABits A;
        A.s[0]  = f2bf(a0.x); A.s[1]  = f2bf(a0.y); A.s[2]  = f2bf(a0.z); A.s[3]  = f2bf(a0.w);
        A.s[4]  = f2bf(a1.x); A.s[5]  = f2bf(a1.y); A.s[6]  = f2bf(a1.z); A.s[7]  = f2bf(a1.w);
        A.s[8]  = f2bf(a2.x); A.s[9]  = f2bf(a2.y); A.s[10] = f2bf(a2.z); A.s[11] = f2bf(a2.w);
        A.s[12] = f2bf(a3.x); A.s[13] = f2bf(a3.y); A.s[14] = f2bf(a3.z); A.s[15] = f2bf(a3.w);

#pragma unroll
        for (int m = 0; m < 3; ++m) {
#pragma unroll
            for (int nt = 0; nt < 4; ++nt) {
                // B operand 32x16: lane col = lq, elem i<8: K = 8h + i,
                // i>=8: K = 16 + 8h + i-8 (same order as A).
                ABits Bv;
                const uint4* wp = (const uint4*)(wt + ((m * kDH + nt * 16 + lq) * kDM
                                                       + kk * 32 + 8 * h));
                Bv.q[0] = wp[0];
                Bv.q[1] = wp[2];
                acc[m][nt] = __builtin_amdgcn_wmma_f32_16x16x32_bf16(
                    false, A.v, false, Bv.v, (short)0, acc[m][nt], false, false);
            }
        }
    }

    // Epilogue: + bias, store. C layout: lane -> (token = r + 8h, col = lq).
    const int bbat  = Mbase >> 11;              // batch index (tiles never straddle)
    const int sbase = (Mbase & (kS - 1)) + 8 * h;
#pragma unroll
    for (int nt = 0; nt < 4; ++nt) {
        const int n   = nt * 16 + lq;
        const float vq = bq[n], vk = bk[n], vv = bv[n];
        union { uint4 q; unsigned short s[8]; } pv;
#pragma unroll
        for (int r = 0; r < 8; ++r) {
            const int token = Mbase + r + 8 * h;
            qb[token * kDH + n] = f2bf(acc[0][nt][r] + vq);
            kb[token * kDH + n] = f2bf(acc[1][nt][r] + vk);
            pv.s[r] = f2bf(acc[2][nt][r] + vv);
        }
        *(uint4*)(vtb + ((bbat * kDH + n) * kS + sbase)) = pv.q;  // 16B aligned
    }
}

// ---------------------------------------------------------------------------
// Kernel 2: flash attention. One wave per 16-query tile, 32-key steps.
//   S^T = K_blk @ q^T  (softmax row == lane&15)
//   O^T = V^T  @ P^T   (P^T re-layouted from C regs with 8 shfl_xor(16))
// ---------------------------------------------------------------------------
__global__ __launch_bounds__(128) void attn_kernel(const unsigned short* __restrict__ qb,
                                                   const unsigned short* __restrict__ kb,
                                                   const unsigned short* __restrict__ vtb,
                                                   float* __restrict__ out) {
    const int lane = threadIdx.x & 31;
    const int lq   = lane & 15;
    const int h    = lane >> 4;
    const int g    = blockIdx.x * 4 + (threadIdx.x >> 5);
    const int b    = g >> 7;                 // 128 q-tiles per batch
    const int qt   = g & 127;
    const int boff = b * (kS * kDH);         // same flat offset for q/k/vT/out

    // q tile as B operand (d-chunk c: elem i -> d = c*32 + 16h + i, contiguous).
    const unsigned short* qrowp = qb + boff + (qt * 16 + lq) * kDH;
    ABits qv[2];
#pragma unroll
    for (int c = 0; c < 2; ++c) {
        const uint4* p = (const uint4*)(qrowp + c * 32 + 16 * h);
        qv[c].q[0] = p[0];
        qv[c].q[1] = p[1];
    }

    v8f o[4];
#pragma unroll
    for (int dt = 0; dt < 4; ++dt)
        o[dt] = (v8f){0.f, 0.f, 0.f, 0.f, 0.f, 0.f, 0.f, 0.f};
    float m_old = -1e30f, lsum = 0.f;
    const int qrow = qt * 16 + lq;           // this lane's softmax row
    const int jmax = (qt * 16 + 15) >> 5;    // causal: last 32-key step

    for (int j = 0; j <= jmax; ++j) {
        const int kvb = j * 32;

        // Prefetch next step's K rows and V lines into WGP$/L2 while this
        // step's WMMA/softmax chain runs (single wave per SIMD -> the only
        // latency hiding we get is what we schedule ourselves).
        if (j < jmax) {
            const unsigned short* kn = kb + boff + (kvb + 32 + lq) * kDH;
            __builtin_prefetch(kn, 0, 3);                 // kv rows +32..+47
            __builtin_prefetch(kn + 16 * kDH, 0, 3);      // kv rows +48..+63
#pragma unroll
            for (int dt = 0; dt < 4; ++dt)
                __builtin_prefetch(vtb + boff + (dt * 16 + lq) * kS + kvb + 32, 0, 3);
        }

        // ---- scores: S^T tiles (16 kv x 16 q), K = 64 over two 32-chunks
        v8f sc[2];
#pragma unroll
        for (int t = 0; t < 2; ++t) {
            sc[t] = (v8f){0.f, 0.f, 0.f, 0.f, 0.f, 0.f, 0.f, 0.f};
            const unsigned short* kr = kb + boff + (kvb + t * 16 + lq) * kDH;
#pragma unroll
            for (int c = 0; c < 2; ++c) {
                ABits Ak;                     // A layout: elem i -> d=c*32+16h+i
                const uint4* p = (const uint4*)(kr + c * 32 + 16 * h);
                Ak.q[0] = p[0];
                Ak.q[1] = p[1];
                sc[t] = __builtin_amdgcn_wmma_f32_16x16x32_bf16(
                    false, Ak.v, false, qv[c].v, (short)0, sc[t], false, false);
            }
        }

        // ---- mask + online softmax (row = lq; stats need one xor-16 hop)
        float pr[2][8];
        float lm = -1e30f;
#pragma unroll
        for (int t = 0; t < 2; ++t)
#pragma unroll
            for (int r = 0; r < 8; ++r) {
                float s = sc[t][r] * 0.125f;                 // 1/sqrt(64)
                const int kv = kvb + t * 16 + r + 8 * h;
                s = (kv > qrow) ? -1e30f : s;
                pr[t][r] = s;
                lm = fmaxf(lm, s);
            }
        const float mcur = fmaxf(lm, __shfl_xor(lm, 16, 32));
        const float mnew = fmaxf(m_old, mcur);
        const float alpha = __expf(m_old - mnew);
        float ls = 0.f;
#pragma unroll
        for (int t = 0; t < 2; ++t)
#pragma unroll
            for (int r = 0; r < 8; ++r) {
                const float e = __expf(pr[t][r] - mnew);
                pr[t][r] = e;
                ls += e;
            }
        ls += __shfl_xor(ls, 16, 32);
        lsum = lsum * alpha + ls;
        m_old = mnew;
#pragma unroll
        for (int dt = 0; dt < 4; ++dt)
#pragma unroll
            for (int r = 0; r < 8; ++r)
                o[dt][r] *= alpha;

        // ---- re-layout P^T (C regs) into WMMA B operand (elem i -> kv=16h+i)
        unsigned pk[2][4], sh[2][4];
#pragma unroll
        for (int t = 0; t < 2; ++t)
#pragma unroll
            for (int jj = 0; jj < 4; ++jj) {
                pk[t][jj] = pack2(pr[t][2 * jj], pr[t][2 * jj + 1]);
                sh[t][jj] = (unsigned)__shfl_xor((int)pk[t][jj], 16, 32);
            }
        ABits Bp;
#pragma unroll
        for (int jj = 0; jj < 4; ++jj) {
            Bp.u[jj]     = h ? sh[1][jj] : pk[0][jj];   // kv 16h + 0..7
            Bp.u[4 + jj] = h ? pk[1][jj] : sh[0][jj];   // kv 16h + 8..15
        }

        // ---- O^T += V^T @ P^T   (A = vT 16d x 32kv, contiguous along kv)
#pragma unroll
        for (int dt = 0; dt < 4; ++dt) {
            const unsigned short* vr = vtb + boff + (dt * 16 + lq) * kS + kvb;
            ABits Av;
            const uint4* p = (const uint4*)(vr + 16 * h);
            Av.q[0] = p[0];
            Av.q[1] = p[1];
            o[dt] = __builtin_amdgcn_wmma_f32_16x16x32_bf16(
                false, Av.v, false, Bp.v, (short)0, o[dt], false, false);
        }
    }

    // ---- normalize + store. O^T layout: lane -> (q = lq, d = dt*16 + r + 8h)
    const float inv = 1.0f / lsum;
    float* obase = out + boff + (qt * 16 + lq) * kDH;
#pragma unroll
    for (int dt = 0; dt < 4; ++dt) {
        float* orow = obase + dt * 16 + 8 * h;
        float4 s0 = {o[dt][0] * inv, o[dt][1] * inv, o[dt][2] * inv, o[dt][3] * inv};
        float4 s1 = {o[dt][4] * inv, o[dt][5] * inv, o[dt][6] * inv, o[dt][7] * inv};
        ((float4*)orow)[0] = s0;
        ((float4*)orow)[1] = s1;
    }
}

// ---------------------------------------------------------------------------
extern "C" void kernel_launch(void* const* d_in, const int* in_sizes, int n_in,
                              void* d_out, int out_size, void* d_ws, size_t ws_size,
                              hipStream_t stream) {
    (void)in_sizes; (void)n_in; (void)out_size; (void)ws_size;

    const float* x  = (const float*)d_in[0];
    const float* Wq = (const float*)d_in[1];
    const float* bq = (const float*)d_in[2];
    const float* Wk = (const float*)d_in[3];
    const float* bk = (const float*)d_in[4];
    const float* Wv = (const float*)d_in[5];
    const float* bv = (const float*)d_in[6];
    float* outp = (float*)d_out;

    char* ws = (char*)d_ws;
    unsigned short* qb  = (unsigned short*)(ws);                        // 1 MB
    unsigned short* kb  = (unsigned short*)(ws + (1u << 20));           // 1 MB
    unsigned short* vtb = (unsigned short*)(ws + (2u << 20));           // 1 MB
    unsigned short* wt  = (unsigned short*)(ws + (3u << 20));           // 384 KB

    // 0) transpose+convert weights: 3*64*1024 elems = 768 blocks of 256
    wt_kernel<<<768, 256, 0, stream>>>(Wq, Wk, Wv, wt);
    // 1) QKV projection: 512 M-tiles, 8 waves/block
    qkv_kernel<<<kMTiles / 8, 256, 0, stream>>>(x, wt, bq, bk, bv, qb, kb, vtb);
    // 2) attention: 512 q-tiles, 4 waves/block
    attn_kernel<<<(kB * (kS / 16)) / 4, 128, 0, stream>>>(qb, kb, vtb, outp);
}
